// MoELayer_10170482557270
// MI455X (gfx1250) — matrix operand order repaired and checked by reading, not compiled, not run on hardware
//
#include <hip/hip_runtime.h>
#include <math.h>

// ---------------------------------------------------------------------------
// Problem constants (from reference): N=8192, D_IN=1024, E=8, H=2048, D_OUT=1024
// ---------------------------------------------------------------------------
#define PN     8192
#define PDIN   1024
#define PE     8
#define PH     2048
#define PDOUT  1024

// Vector types
typedef __attribute__((ext_vector_type(16))) __bf16          v16bf;
typedef __attribute__((ext_vector_type(8)))  float           v8f;
typedef __attribute__((ext_vector_type(4)))  unsigned int    v4u;
typedef __attribute__((ext_vector_type(4)))  int             v4i;
typedef __attribute__((ext_vector_type(4)))  float           v4f;
typedef __attribute__((ext_vector_type(4)))  unsigned short  v4us;

// Address-space-qualified pointer types for the async global->LDS builtin
typedef __attribute__((address_space(1))) v4i* gbl_v4i_ptr;
typedef __attribute__((address_space(3))) v4i* lds_v4i_ptr;

union FragBF {
    v16bf v;
    v4u   q[2];
};

// ---------------------------------------------------------------------------
// CDNA5 async global->LDS staging (ASYNCcnt path), with register-path fallback
// ---------------------------------------------------------------------------
#if defined(__has_builtin)
#  if __has_builtin(__builtin_amdgcn_global_load_async_to_lds_b128) && \
      __has_builtin(__builtin_amdgcn_s_wait_asynccnt)
#    define USE_ASYNC_LDS 1
#  endif
#endif
#ifndef USE_ASYNC_LDS
#  define USE_ASYNC_LDS 0
#endif

__device__ __forceinline__ void stage_b128(const unsigned short* g, unsigned short* l)
{
#if USE_ASYNC_LDS
    __builtin_amdgcn_global_load_async_to_lds_b128(
        (gbl_v4i_ptr)const_cast<unsigned short*>(g),
        (lds_v4i_ptr)l,
        0, 0);
#else
    *(v4u*)l = *(const v4u*)g;
#endif
}

__device__ __forceinline__ void wait_stage()
{
#if USE_ASYNC_LDS
    __builtin_amdgcn_s_wait_asynccnt(0);
#endif
}

// float -> bf16 (round-to-nearest-even) as raw ushort
__device__ __forceinline__ unsigned short f2bf(float f) {
    unsigned int u = __float_as_uint(f);
    u += 0x7fffu + ((u >> 16) & 1u);
    return (unsigned short)(u >> 16);
}

// ---------------------------------------------------------------------------
// Gating kernel: gates[n,:] = softmax(x[n,:] @ Wg + bg)   (one wave per row)
// ---------------------------------------------------------------------------
__global__ __launch_bounds__(256) void gates_kernel(
    const float* __restrict__ x, const float* __restrict__ Wg,
    const float* __restrict__ bg, float* __restrict__ gates)
{
    const int lane = threadIdx.x & 31;
    const int wid  = threadIdx.x >> 5;
    const int row  = blockIdx.x * 8 + wid;

    float p[PE];
#pragma unroll
    for (int e = 0; e < PE; ++e) p[e] = 0.f;

    const float* xr = x + (size_t)row * PDIN;
    for (int i = lane; i < PDIN; i += 32) {
        float xv = xr[i];
        const v4f* w = (const v4f*)&Wg[(size_t)i * PE];
        v4f w0 = w[0];
        v4f w1 = w[1];
        p[0] += xv * w0[0]; p[1] += xv * w0[1];
        p[2] += xv * w0[2]; p[3] += xv * w0[3];
        p[4] += xv * w1[0]; p[5] += xv * w1[1];
        p[6] += xv * w1[2]; p[7] += xv * w1[3];
    }
#pragma unroll
    for (int off = 16; off > 0; off >>= 1) {
#pragma unroll
        for (int e = 0; e < PE; ++e) p[e] += __shfl_xor(p[e], off, 32);
    }
    if (lane == 0) {
        float lg[PE], mx = -3.0e38f;
#pragma unroll
        for (int e = 0; e < PE; ++e) { lg[e] = p[e] + bg[e]; mx = fmaxf(mx, lg[e]); }
        float s = 0.f;
#pragma unroll
        for (int e = 0; e < PE; ++e) { lg[e] = __expf(lg[e] - mx); s += lg[e]; }
        float inv = 1.f / s;
#pragma unroll
        for (int e = 0; e < PE; ++e) gates[(size_t)row * PE + e] = lg[e] * inv;
    }
}

// ---------------------------------------------------------------------------
// Elementwise fp32 -> bf16 conversion (4 elements/thread)
// ---------------------------------------------------------------------------
__global__ __launch_bounds__(256) void convert_bf16_kernel(
    const float* __restrict__ src, unsigned short* __restrict__ dst)
{
    const size_t i = (size_t)blockIdx.x * 256 + threadIdx.x;
    v4f v = ((const v4f*)src)[i];
    v4us o;
    o[0] = f2bf(v[0]); o[1] = f2bf(v[1]); o[2] = f2bf(v[2]); o[3] = f2bf(v[3]);
    ((v4us*)dst)[i] = o;
}

// ---------------------------------------------------------------------------
// Tiled transpose + fp32->bf16:  src[R][C] -> dst[C][R]  (32x32 tiles via LDS)
// ---------------------------------------------------------------------------
__global__ __launch_bounds__(256) void transpose_bf16_kernel(
    const float* __restrict__ src, unsigned short* __restrict__ dst,
    int R, int C)
{
    __shared__ unsigned short tile[32][33];
    const int tx = threadIdx.x & 31;
    const int ty = threadIdx.x >> 5;           // 0..7
    const int r0 = blockIdx.y * 32;
    const int c0 = blockIdx.x * 32;

#pragma unroll
    for (int p = 0; p < 4; ++p) {
        int r = ty + p * 8;
        tile[r][tx] = f2bf(src[(size_t)(r0 + r) * C + c0 + tx]);
    }
    __syncthreads();
#pragma unroll
    for (int p = 0; p < 4; ++p) {
        int rr = ty + p * 8;
        dst[(size_t)(c0 + rr) * R + r0 + tx] = tile[tx][rr];
    }
}

// ---------------------------------------------------------------------------
// bf16 WMMA GEMM: C[M,N] = A[M,K] * B^T[N,K]  (both operands row-major bf16)
// Block tile 128x128, K-step 64, 8 waves (4 in M x 2 in N), each wave 2x4
// 16x16x32 bf16 WMMA accumulators. LDS double-buffered; tiles staged with
// CDNA5 async global->LDS loads (ASYNCcnt) when available.
//   EPI==0: Hout = relu(C + bias)             (bf16, stride PH)
//   EPI==1: Fout (+)= gates[:,e] * (C + bias) (fp32, stride PDOUT)
// ---------------------------------------------------------------------------
#define TM 128
#define TN 128
#define TK 64
#define LDSS 72   // ushort stride (144 bytes) to break LDS bank conflicts

template <int KTOT, int LDA, int LDB, int EPI>
__global__ __launch_bounds__(256) void gemm_bf16_kernel(
    const unsigned short* __restrict__ A,
    const unsigned short* __restrict__ B,
    const float* __restrict__ bias,
    const float* __restrict__ gates, int expert, int accflag,
    unsigned short* __restrict__ Hout,
    float* __restrict__ Fout)
{
    __shared__ unsigned short sA[2][TM * LDSS];
    __shared__ unsigned short sB[2][TN * LDSS];
    __shared__ float gls[TM];

    const int tid    = threadIdx.x;
    const int lane   = tid & 31;
    const int wid    = tid >> 5;
    const int waveM  = wid & 3;     // 0..3 -> 32 rows each
    const int waveN  = wid >> 2;    // 0..1 -> 64 cols each
    const int lh     = lane >> 4;   // lane half
    const int lm     = lane & 15;
    const int blockM = blockIdx.y * TM;
    const int blockN = blockIdx.x * TN;

    if (EPI == 1) {
        if (tid < TM) gls[tid] = gates[(size_t)(blockM + tid) * PE + expert];
    }

    // Stage one 128x64 A tile + one 128x64 B tile into LDS buffer `buf`.
    auto stageAB = [&](int buf, int kb) {
#pragma unroll
        for (int p = 0; p < 4; ++p) {
            int idx = p * 256 + tid;
            int row = idx >> 3;
            int cg  = idx & 7;
            stage_b128(&A[(size_t)(blockM + row) * LDA + kb + cg * 8],
                       &sA[buf][row * LDSS + cg * 8]);
            stage_b128(&B[(size_t)(blockN + row) * LDB + kb + cg * 8],
                       &sB[buf][row * LDSS + cg * 8]);
        }
    };

    v8f acc[2][4] = {};

    stageAB(0, 0);
    const int NIT = KTOT / TK;
    for (int it = 0; it < NIT; ++it) {
        const int cur = it & 1;
        wait_stage();          // this wave's tile data landed in LDS
        __syncthreads();       // everyone's landed; prev buffer fully consumed
        if (it + 1 < NIT) stageAB(cur ^ 1, (it + 1) * TK);  // overlap with compute

        const unsigned short* cA = sA[cur];
        const unsigned short* cB = sB[cur];

#pragma unroll
        for (int s = 0; s < 2; ++s) {   // two k=32 sub-steps
            FragBF a[2], b[4];
            // A fragments: lanes 0-15 rows M, K interleave {0..7,16..23}/{8..15,24..31}
#pragma unroll
            for (int mi = 0; mi < 2; ++mi) {
                int row = waveM * 32 + mi * 16 + lm;
                const unsigned short* base = &cA[row * LDSS + s * 32 + lh * 8];
                a[mi].q[0] = *(const v4u*)base;
                a[mi].q[1] = *(const v4u*)(base + 16);
            }
            // B fragments: same K interleave as A {0..7,16..23}/{8..15,24..31}
#pragma unroll
            for (int ni = 0; ni < 4; ++ni) {
                int col = waveN * 64 + ni * 16 + lm;
                const unsigned short* base = &cB[col * LDSS + s * 32 + lh * 8];
                b[ni].q[0] = *(const v4u*)base;
                b[ni].q[1] = *(const v4u*)(base + 16);
            }
#pragma unroll
            for (int mi = 0; mi < 2; ++mi)
#pragma unroll
                for (int ni = 0; ni < 4; ++ni)
                    acc[mi][ni] = __builtin_amdgcn_wmma_f32_16x16x32_bf16(
                        false, a[mi].v, false, b[ni].v,
                        (short)0, acc[mi][ni], false, false);
        }
    }

    // Epilogue. C layout: VGPR v, lanes 0-15 -> M=v, lanes 16-31 -> M=v+8; N=lm
#pragma unroll
    for (int mi = 0; mi < 2; ++mi) {
        int row0 = blockM + waveM * 32 + mi * 16;
#pragma unroll
        for (int ni = 0; ni < 4; ++ni) {
            int col  = blockN + waveN * 64 + ni * 16 + lm;
            float bv = bias[col];
#pragma unroll
            for (int v = 0; v < 8; ++v) {
                int   r   = row0 + v + 8 * lh;
                float val = acc[mi][ni][v] + bv;
                if (EPI == 0) {
                    val = val > 0.f ? val : 0.f;
                    Hout[(size_t)r * PH + col] = f2bf(val);
                } else {
                    float o = gls[r - blockM] * val;
                    size_t idx = (size_t)r * PDOUT + col;
                    if (accflag) o += Fout[idx];
                    Fout[idx] = o;
                }
            }
        }
    }
}

// ---------------------------------------------------------------------------
// Host-side launch
// ---------------------------------------------------------------------------
extern "C" void kernel_launch(void* const* d_in, const int* in_sizes, int n_in,
                              void* d_out, int out_size, void* d_ws, size_t ws_size,
                              hipStream_t stream)
{
    const float* x  = (const float*)d_in[0];
    const float* Wg = (const float*)d_in[1];
    const float* bg = (const float*)d_in[2];
    const float* W1 = (const float*)d_in[3];
    const float* b1 = (const float*)d_in[4];
    const float* W2 = (const float*)d_in[5];
    const float* b2 = (const float*)d_in[6];
    float* out = (float*)d_out;

    // Workspace carve-up (all offsets 16B aligned)
    char* ws = (char*)d_ws;
    float* gates = (float*)ws;                 ws += (size_t)PN * PE * sizeof(float);  // 256 KB
    unsigned short* xb  = (unsigned short*)ws; ws += (size_t)PN * PDIN * 2;            // 16 MB
    unsigned short* w1t = (unsigned short*)ws; ws += (size_t)PH * PDIN * 2;            // 4 MB
    unsigned short* w2t = (unsigned short*)ws; ws += (size_t)PDOUT * PH * 2;           // 4 MB
    unsigned short* hb  = (unsigned short*)ws;                                         // 32 MB

    // 1) gates = softmax(x @ Wg + bg)
    gates_kernel<<<PN / 8, 256, 0, stream>>>(x, Wg, bg, gates);

    // 2) x -> bf16
    convert_bf16_kernel<<<(PN * PDIN) / (256 * 4), 256, 0, stream>>>(x, xb);

    // 3) Per-expert: transpose weights to bf16, GEMM1 (relu), GEMM2 (gated acc)
    for (int e = 0; e < PE; ++e) {
        // W1[e]: [D_IN, H] -> w1t: [H, D_IN]
        transpose_bf16_kernel<<<dim3(PH / 32, PDIN / 32), 256, 0, stream>>>(
            W1 + (size_t)e * PDIN * PH, w1t, PDIN, PH);
        // W2[e]: [H, D_OUT] -> w2t: [D_OUT, H]
        transpose_bf16_kernel<<<dim3(PDOUT / 32, PH / 32), 256, 0, stream>>>(
            W2 + (size_t)e * PH * PDOUT, w2t, PH, PDOUT);

        // h = relu(xb @ W1[e] + b1[e])      M=8192 N=2048 K=1024
        gemm_bf16_kernel<PDIN, PDIN, PDIN, 0>
            <<<dim3(PH / TN, PN / TM), 256, 0, stream>>>(
                xb, w1t, b1 + (size_t)e * PH,
                nullptr, 0, 0, hb, nullptr);

        // out (+)= gates[:,e] * (h @ W2[e] + b2[e])   M=8192 N=1024 K=2048
        gemm_bf16_kernel<PH, PH, PH, 1>
            <<<dim3(PDOUT / TN, PN / TM), 256, 0, stream>>>(
                hb, w2t, b2 + (size_t)e * PDOUT,
                gates, e, (e > 0) ? 1 : 0, nullptr, out);
    }
}
